// ScaledDotProductAttention_85478439125336
// MI455X (gfx1250) — matrix operand order repaired and checked by reading, not compiled, not run on hardware
//
#include <hip/hip_runtime.h>
#include <math.h>
#include <stdint.h>

typedef _Float16 f16;
typedef __attribute__((ext_vector_type(16))) _Float16 v16h;
typedef __attribute__((ext_vector_type(8)))  _Float16 v8h;
typedef __attribute__((ext_vector_type(8)))  float    v8f;
typedef __attribute__((ext_vector_type(4)))  float    v4f;
typedef __attribute__((ext_vector_type(2)))  float    v2f;
typedef __attribute__((ext_vector_type(4)))  int      v4i;

#define BH 16
#define Lq 2048
#define Dd 64
#define QT 64            // q rows per workgroup
#define KT 32            // k cols per tile
#define NW 4             // waves per workgroup
#define KROW (Dd + 4)    // f32 K-tile row stride (68 floats = 272B, 16B aligned, bank-spread)
#define VROW (KT + 8)    // f16 V-tile row stride (40 halves = 80B)

union V16u { v16h v; v8h h[2]; };

// ---- CDNA5 async global->LDS 16B copy (ASYNCcnt path) ----
__device__ __forceinline__ void async_cp16(const void* g, void* l) {
#if __has_builtin(__builtin_amdgcn_global_load_async_to_lds_b128)
  __builtin_amdgcn_global_load_async_to_lds_b128(
      (__attribute__((address_space(1))) v4i*)(uintptr_t)g,
      (__attribute__((address_space(3))) v4i*)(uint32_t)(uintptr_t)l,
      0, 0);
#else
  uint32_t lo = (uint32_t)(uintptr_t)l;
  asm volatile("global_load_async_to_lds_b128 %0, %1, off"
               :: "v"(lo), "v"(g) : "memory");
#endif
}
__device__ __forceinline__ void wait_async0() {
#if __has_builtin(__builtin_amdgcn_s_wait_asynccnt)
  __builtin_amdgcn_s_wait_asynccnt(0);
#else
  asm volatile("s_wait_asynccnt 0x0" ::: "memory");
#endif
}

__device__ __forceinline__ float rmax16(float v) {
  v = fmaxf(v, __shfl_xor(v, 1, 32));
  v = fmaxf(v, __shfl_xor(v, 2, 32));
  v = fmaxf(v, __shfl_xor(v, 4, 32));
  v = fmaxf(v, __shfl_xor(v, 8, 32));
  return v;
}
__device__ __forceinline__ float rsum16(float v) {
  v += __shfl_xor(v, 1, 32);
  v += __shfl_xor(v, 2, 32);
  v += __shfl_xor(v, 4, 32);
  v += __shfl_xor(v, 8, 32);
  return v;
}

__global__ __launch_bounds__(128)
void realformer_fa(const float* __restrict__ Qg, const float* __restrict__ Kg,
                   const float* __restrict__ Vg, const float* __restrict__ PSin,
                   float* __restrict__ ctxOut, float* __restrict__ scoresOut)
{
  // Double-buffered tiles: K as raw f32 (filled by async DMA), V as f16 transposed.
  __shared__ __align__(16) float    Ks32[2][KT][KROW];     // [k][d] f32
  __shared__ __align__(16) _Float16 Vt[2][Dd][VROW];       // [d][k] f16
  __shared__ __align__(16) _Float16 Ps[NW][16][VROW];      // per-wave P tile [m][k]

  const int tid  = threadIdx.x;
  const int lane = tid & 31;
  const int wave = tid >> 5;
  const int g    = lane >> 4;     // half-wave group
  const int ln   = lane & 15;
  const int bh   = blockIdx.y;
  const int qBase = blockIdx.x * QT;
  const int q0w  = qBase + wave * 16;

  const size_t rowBase = (size_t)bh * Lq;

  // ---- issue async K-tile copy: 8KB tile, 4 x b128 per thread ----
  auto issueK = [&](int kb, int buf) {
    const char* gbase = (const char*)(Kg + (rowBase + kb) * Dd);
    #pragma unroll
    for (int j = 0; j < 4; ++j) {
      int o  = tid * 16 + j * 2048;   // byte offset within 8KB tile
      int r  = o >> 8;                // 256B per data row
      int cB = o & 255;
      async_cp16(gbase + o, (char*)&Ks32[buf][r][0] + cB);
    }
  };
  // ---- manual V tile load: f32 -> f16, transposed to [d][k] ----
  auto loadV = [&](int kb, int buf) {
    for (int i = tid; i < KT * Dd; i += 128) {
      int r = i >> 6, c = i & 63;
      Vt[buf][c][r] = (f16)Vg[(rowBase + kb + r) * Dd + c];
    }
  };

  // ---- Q A-operands in contiguous-k order (same as K B-operand) (16x32 x2 d-chunks), 1/sqrt(D) folded.
  //      K = (lane>=16?16:0) + v*2 + j ; row M = lane&15.
  V16u qa[2];
  {
    const float* qrow = Qg + (rowBase + q0w + ln) * Dd;
    #pragma unroll
    for (int c = 0; c < 2; ++c) {
      #pragma unroll
      for (int v = 0; v < 8; ++v) {
        int dbase = c * 32 + g * 16 + v * 2;
        v2f q2 = *(const v2f*)(qrow + dbase);
        qa[c].v[2 * v]     = (f16)(q2.x * 0.125f);
        qa[c].v[2 * v + 1] = (f16)(q2.y * 0.125f);
      }
    }
  }

  v8f ctx[4];
  #pragma unroll
  for (int n = 0; n < 4; ++n) ctx[n] = (v8f)0.0f;
  float mrow[8], lrow[8];
  #pragma unroll
  for (int r = 0; r < 8; ++r) { mrow[r] = -INFINITY; lrow[r] = 0.0f; }

  const int nkt = (qBase + QT) / KT;   // >= 2 always

  // ---- software pipeline prologue ----
  issueK(0, 0);
  loadV(0, 0);
  int bsel = 0;

  for (int kt = 0; kt < nkt; ++kt) {
    const int kb = kt * KT;
    wait_async0();        // K[kt] DMA complete (this wave's copies)
    __syncthreads();      // all waves' K DMA + V stores visible

    const bool more = (kt + 1 < nkt);
    if (more) {
      issueK(kb + KT, bsel ^ 1);   // overlap next K DMA with compute
      __builtin_prefetch(PSin + ((size_t)bh * Lq + q0w + ln) * Lq + kb + KT + g * 16, 0, 1);
    }

    // ---- S = Q*K^T : B-operand built from f32 LDS with on-the-fly f16 convert.
    //      B layout: lane holds col N=lane&15; group g selects contraction half;
    //      VGPR v half j holds K = g*16 + 2v + j (16 contiguous floats in LDS).
    v8f s[2];
    #pragma unroll
    for (int n = 0; n < 2; ++n) {
      v8f acc = (v8f)0.0f;
      #pragma unroll
      for (int c = 0; c < 2; ++c) {
        const float* kr32 = &Ks32[bsel][n * 16 + ln][c * 32 + g * 16];
        v4f a0 = *(const v4f*)(kr32 + 0);
        v4f a1 = *(const v4f*)(kr32 + 4);
        v4f a2 = *(const v4f*)(kr32 + 8);
        v4f a3 = *(const v4f*)(kr32 + 12);
        V16u kbop;
        #pragma unroll
        for (int t = 0; t < 4; ++t) {
          kbop.v[t]      = (f16)a0[t];
          kbop.v[4 + t]  = (f16)a1[t];
          kbop.v[8 + t]  = (f16)a2[t];
          kbop.v[12 + t] = (f16)a3[t];
        }
        acc = __builtin_amdgcn_wmma_f32_16x16x32_f16(
                  false, qa[c].v, false, kbop.v, (short)0, acc, false, false);
      }
      s[n] = acc;
    }

    // ---- add prev_scores, causal mask, emit score tile, row max ----
    float tmax[8];
    #pragma unroll
    for (int r = 0; r < 8; ++r) {
      const int q = q0w + r + 8 * g;
      const size_t rowOff = ((size_t)bh * Lq + q) * (size_t)Lq + kb + ln;
      float s0 = s[0][r] + PSin[rowOff];
      float s1 = s[1][r] + PSin[rowOff + 16];
      if (kb + ln      > q) s0 = -INFINITY;
      if (kb + 16 + ln > q) s1 = -INFINITY;
      scoresOut[rowOff]      = s0;
      scoresOut[rowOff + 16] = s1;
      s[0][r] = s0;
      s[1][r] = s1;
      tmax[r] = rmax16(fmaxf(s0, s1));
    }

    // ---- online softmax; stage P (f16) into per-wave LDS scratch ----
    #pragma unroll
    for (int r = 0; r < 8; ++r) {
      float mn = fmaxf(mrow[r], tmax[r]);
      float alpha = __expf(mrow[r] - mn);   // first tile: exp(-inf - m) = 0
      float p0 = __expf(s[0][r] - mn);      // masked -> exp(-inf) = 0
      float p1 = __expf(s[1][r] - mn);
      lrow[r] = lrow[r] * alpha + rsum16(p0 + p1);
      mrow[r] = mn;
      #pragma unroll
      for (int n = 0; n < 4; ++n) ctx[n][r] *= alpha;
      Ps[wave][r + 8 * g][ln]      = (f16)p0;
      Ps[wave][r + 8 * g][16 + ln] = (f16)p1;
    }

    // ---- reload P in A layout (same-wave LDS ops in-order) ----
    V16u pa;
    {
      const f16* pr = &Ps[wave][ln][g * 8];
      pa.h[0] = *(const v8h*)pr;          // K = g*8 + 0..7
      pa.h[1] = *(const v8h*)(pr + 16);   // K = 16 + g*8 + 0..7
    }

    // ---- ctx += P * V ----
    #pragma unroll
    for (int n = 0; n < 4; ++n) {
      V16u vb;
      const f16* vr = &Vt[bsel][n * 16 + ln][g * 8];
      vb.h[0] = *(const v8h*)vr;
      vb.h[1] = *(const v8h*)(vr + 16);
      ctx[n] = __builtin_amdgcn_wmma_f32_16x16x32_f16(
                   false, pa.v, false, vb.v, (short)0, ctx[n], false, false);
    }

    // ---- software-prefetch next V tile into the other buffer ----
    if (more) loadV(kb + KT, bsel ^ 1);
    bsel ^= 1;
  }

  // ---- finalize context ----
  #pragma unroll
  for (int r = 0; r < 8; ++r) {
    const int q = q0w + r + 8 * g;
    const float inv = 1.0f / lrow[r];
    float* crow = ctxOut + (rowBase + q) * Dd;
    #pragma unroll
    for (int n = 0; n < 4; ++n)
      crow[n * 16 + ln] = ctx[n][r] * inv;
  }

  // ---- fill strictly-future score columns with -inf (rectangular region) ----
  const int kStart = qBase + QT;
  if (kStart < Lq) {
    const int width = Lq - kStart;
    const float4 ninf = make_float4(-INFINITY, -INFINITY, -INFINITY, -INFINITY);
    for (int r = 0; r < QT; ++r) {
      float* dst = scoresOut + ((size_t)bh * Lq + qBase + r) * (size_t)Lq + kStart;
      for (int c = tid * 4; c < width; c += 128 * 4)
        *(float4*)(dst + c) = ninf;
    }
  }
}

extern "C" void kernel_launch(void* const* d_in, const int* in_sizes, int n_in,
                              void* d_out, int out_size, void* d_ws, size_t ws_size,
                              hipStream_t stream) {
  (void)in_sizes; (void)n_in; (void)d_ws; (void)ws_size; (void)out_size;
  const float* Q  = (const float*)d_in[0];
  const float* K  = (const float*)d_in[1];
  const float* V  = (const float*)d_in[2];
  const float* PS = (const float*)d_in[3];
  float* ctxOut    = (float*)d_out;                      // [B,H,L,D]
  float* scoresOut = ctxOut + (size_t)BH * Lq * Dd;      // [B,H,L,L]
  dim3 grid(Lq / QT, BH);
  realformer_fa<<<grid, 128, 0, stream>>>(Q, K, V, PS, ctxOut, scoresOut);
}
